// EvoformerBlock_40501541601305
// MI455X (gfx1250) — matrix operand order repaired and checked by reading, not compiled, not run on hardware
//
#include <hip/hip_runtime.h>
#include <math.h>

typedef unsigned short h16;                    // opaque f16 storage for host code
typedef __attribute__((ext_vector_type(16))) _Float16 v16h;
typedef __attribute__((ext_vector_type(8)))  _Float16 v8h;
typedef __attribute__((ext_vector_type(8)))  float    v8f;
typedef __attribute__((ext_vector_type(4)))  float    f4v;

#define S_  128
#define N_  256
#define CM  256
#define CZ  128
#define CT  128
#define CO  32
#define HM  8
#define DM  32
#define HP  4
#define DP  32
#define NEGB 1e9f

// Input tensor indices (setup_inputs() dict insertion order, recursively flattened)
enum {
  IN_M = 0, IN_Z, IN_MSAMASK, IN_PAIRMASK,
  R_LN_G, R_LN_B, R_QW, R_KW, R_VW, R_GW, R_GB, R_OW, R_OB, R_LNZ_G, R_LNZ_B, R_BW,  // 4..15
  C_LN_G, C_LN_B, C_QW, C_KW, C_VW, C_GW, C_GB, C_OW, C_OB,                          // 16..24
  MT_LN_G, MT_LN_B, MT_L1W, MT_L1B, MT_L2W, MT_L2B,                                  // 25..30
  OP_LN_G, OP_LN_B, OP_AW, OP_AB, OP_BW, OP_BB, OP_OW, OP_OB,                        // 31..38
  TMO_BASE = 39,   // ln.g ln.b ag.w ag.b ap.w ap.b bg.w bg.b bp.w bp.b g.w g.b lno.g lno.b out.w out.b
  TMI_BASE = 55,
  TAS_BASE = 71,   // ln.g ln.b q.w k.w v.w gate.w gate.b out.w out.b bias.w
  TAE_BASE = 81,
  PT_LN_G = 91, PT_LN_B, PT_L1W, PT_L1B, PT_L2W, PT_L2B
};

// ---------------------------------------------------------------------------
// Fragment loaders (layouts per CDNA5 ISA §7.12.2, wave32)
// A 16x32 fragment: lane lo = row; halves 0..7 -> K kb..kb+7,
// halves 8..15 -> K kb+16..kb+23, kb = k0 (+8 for hi lanes).
// B 32x16 fragment: lane lo = col n; halves 0..7 -> K kb..kb+7, 8..15 -> K kb+16..kb+23 (kb +8 hi).
// ---------------------------------------------------------------------------
__device__ __forceinline__ v16h a_frag_f32(const float* __restrict__ Ab, int lda,
                                           int row, int k0, int hi)
{
  const float* ap = Ab + (long)row * lda + k0 + (hi ? 8 : 0);
  __builtin_prefetch(ap + 32, 0, 3);     // next K tile -> global_prefetch_b8
  f4v a0 = *(const f4v*)(ap);
  f4v a1 = *(const f4v*)(ap + 4);
  f4v a2 = *(const f4v*)(ap + 16);
  f4v a3 = *(const f4v*)(ap + 20);
  v16h af;
#pragma unroll
  for (int e = 0; e < 4; ++e) {
    af[e]      = (_Float16)a0[e];
    af[4 + e]  = (_Float16)a1[e];
    af[8 + e]  = (_Float16)a2[e];
    af[12 + e] = (_Float16)a3[e];
  }
  return af;
}

__device__ __forceinline__ v16h a_frag_f16(const h16* __restrict__ Ahb, int lda,
                                           int row, int k0, int hi)
{
  const _Float16* ap = (const _Float16*)Ahb + (long)row * lda + k0 + (hi ? 8 : 0);
  __builtin_prefetch(ap + 32, 0, 3);
  v8h a0 = *(const v8h*)(ap);
  v8h a1 = *(const v8h*)(ap + 16);
  v16h af;
#pragma unroll
  for (int e = 0; e < 8; ++e) { af[e] = a0[e]; af[8 + e] = a1[e]; }
  return af;
}

template <int GUARD>
__device__ __forceinline__ v16h b_frag16(const h16* __restrict__ Bhb, int ldb,
                                         int N, int bn, int kb)
{
  v16h bf;
  if (!GUARD || bn < N) {
    const _Float16* bp = (const _Float16*)Bhb + (long)bn * ldb + kb;
    v8h b0 = *(const v8h*)(bp);
    v8h b1 = *(const v8h*)(bp + 16);
#pragma unroll
    for (int e = 0; e < 8; ++e) { bf[e] = b0[e]; bf[8 + e] = b1[e]; }
  } else {
#pragma unroll
    for (int e = 0; e < 16; ++e) bf[e] = (_Float16)0.f;
  }
  return bf;
}

template <int GUARD>
__device__ __forceinline__ void c_store(float* __restrict__ Cb, h16* __restrict__ C16b,
                                        v8f acc, int row0, int col, int N, int ldc,
                                        const float* __restrict__ bias,
                                        float alpha, int act, int beta, int hi)
{
  if (!GUARD || col < N) {
    float bv = bias ? bias[col] : 0.f;
#pragma unroll
    for (int r = 0; r < 8; ++r) {
      int row = row0 + r + (hi ? 8 : 0);
      float v = acc[r] * alpha + bv;
      if (act == 1)      v = v > 0.f ? v : 0.f;
      else if (act == 2) v = 1.f / (1.f + __expf(-v));
      long off = (long)row * ldc + col;
      if (beta) v += Cb[off];
      Cb[off] = v;
      if (C16b) ((_Float16*)C16b)[off] = (_Float16)v;   // f16 mirror
    }
  }
}

// ---------------------------------------------------------------------------
// Batched GEMM: C = act(alpha*A*B + bias) [+C if beta], optional f16 mirror.
// A: f32 row-major (AF16=0) or f16 row-major (AF16=1).  B: f16 NxK (ldb=K-row).
// GUARD=1 only when N%32!=0 (tiny bias projections); hot GEMMs are unguarded.
// 32x32 tile per wave, 4 v_wmma_f32_16x16x32_f16 per K-step.
// Assumes M%32==0, K%32==0.
// Batch offset = (b / batchH)*s?1 + (b % batchH)*s?2 (elements of own type).
// ---------------------------------------------------------------------------
template <int AF16, int GUARD>
__global__ void __launch_bounds__(128)
gemm_wmma_k(const float* __restrict__ A, const h16* __restrict__ Ah,
            const h16* __restrict__ Bh, const float* __restrict__ bias,
            float* __restrict__ C, h16* __restrict__ C16,
            int M, int N, int K, int lda, int ldb, int ldc, int batchH,
            long sA1, long sA2, long sB1, long sB2, long sC1, long sC2,
            float alpha, int act, int beta)
{
  int b  = blockIdx.y;
  int b1 = b / batchH, b2 = b - b1 * batchH;
  long offA = (long)b1 * sA1 + (long)b2 * sA2;
  long offC = (long)b1 * sC1 + (long)b2 * sC2;
  const float* Ab  = AF16 ? nullptr : A + offA;
  const h16*   Ahb = AF16 ? Ah + offA : nullptr;
  const h16*   Bhb = Bh + (long)b1 * sB1 + (long)b2 * sB2;
  float*       Cb  = C + offC;
  h16*         C16b = C16 ? C16 + offC : nullptr;

  int tN = (N + 31) >> 5;
  int tM = M >> 5;
  int wv = blockIdx.x * 4 + (threadIdx.x >> 5);
  if (wv >= tM * tN) return;
  int m0 = (wv / tN) << 5;
  int n0 = (wv % tN) << 5;
  int lane = threadIdx.x & 31;
  int lo = lane & 15;
  int hi = lane >> 4;

  v8f acc00 = {}, acc01 = {}, acc10 = {}, acc11 = {};
  for (int k0 = 0; k0 < K; k0 += 32) {
    v16h a0 = AF16 ? a_frag_f16(Ahb, lda, m0 + lo, k0, hi)
                   : a_frag_f32(Ab, lda, m0 + lo, k0, hi);
    v16h a1 = AF16 ? a_frag_f16(Ahb, lda, m0 + 16 + lo, k0, hi)
                   : a_frag_f32(Ab, lda, m0 + 16 + lo, k0, hi);
    int kb = k0 + (hi ? 8 : 0);
    v16h b0 = b_frag16<GUARD>(Bhb, ldb, N, n0 + lo, kb);
    v16h b1 = b_frag16<GUARD>(Bhb, ldb, N, n0 + 16 + lo, kb);
    acc00 = __builtin_amdgcn_wmma_f32_16x16x32_f16(false, a0, false, b0, (short)0, acc00, false, false);
    acc01 = __builtin_amdgcn_wmma_f32_16x16x32_f16(false, a0, false, b1, (short)0, acc01, false, false);
    acc10 = __builtin_amdgcn_wmma_f32_16x16x32_f16(false, a1, false, b0, (short)0, acc10, false, false);
    acc11 = __builtin_amdgcn_wmma_f32_16x16x32_f16(false, a1, false, b1, (short)0, acc11, false, false);
  }
  c_store<GUARD>(Cb, C16b, acc00, m0,      n0 + lo,      N, ldc, bias, alpha, act, beta, hi);
  c_store<GUARD>(Cb, C16b, acc01, m0,      n0 + 16 + lo, N, ldc, bias, alpha, act, beta, hi);
  c_store<GUARD>(Cb, C16b, acc10, m0 + 16, n0 + lo,      N, ldc, bias, alpha, act, beta, hi);
  c_store<GUARD>(Cb, C16b, acc11, m0 + 16, n0 + 16 + lo, N, ldc, bias, alpha, act, beta, hi);
}

// ---------------------------------------------------------------------------
// LayerNorm over last dim C, one block (128 threads) per row; in-place safe.
// ---------------------------------------------------------------------------
__global__ void __launch_bounds__(128)
ln_k(const float* __restrict__ x, float* __restrict__ y,
     const float* __restrict__ g, const float* __restrict__ b, int C)
{
  long row = blockIdx.x;
  const float* xr = x + row * C;
  float* yr = y + row * C;
  __shared__ float red[4];
  int t = threadIdx.x;
  float s = 0.f;
  for (int c = t; c < C; c += 128) s += xr[c];
  for (int o = 16; o; o >>= 1) s += __shfl_down(s, o, 32);
  if ((t & 31) == 0) red[t >> 5] = s;
  __syncthreads();
  float mu = (red[0] + red[1] + red[2] + red[3]) / (float)C;
  __syncthreads();
  float v = 0.f;
  for (int c = t; c < C; c += 128) { float d = xr[c] - mu; v += d * d; }
  for (int o = 16; o; o >>= 1) v += __shfl_down(v, o, 32);
  if ((t & 31) == 0) red[t >> 5] = v;
  __syncthreads();
  float var = (red[0] + red[1] + red[2] + red[3]) / (float)C;
  float rs = rsqrtf(var + 1e-5f);
  for (int c = t; c < C; c += 128) yr[c] = (xr[c] - mu) * rs * g[c] + b[c];
}

// ---------------------------------------------------------------------------
// Softmax over rows of length L (128/256). Row r -> (sI, h, i).
// ---------------------------------------------------------------------------
__global__ void __launch_bounds__(128)
softmax_k(float* __restrict__ x, int L, int Mrows, int H,
          const float* __restrict__ bias, int biasH,
          const float* __restrict__ mask, int mldr, int mldj)
{
  long r = blockIdx.x;
  int i  = (int)(r % Mrows);
  int h  = (int)((r / Mrows) % H);
  int sI = (int)(r / ((long)Mrows * H));
  float* xr = x + r * L;
  int t = threadIdx.x;
  __shared__ float red[4];
  float vals[2];
  float mx = -3.0e38f;
#pragma unroll
  for (int p = 0; p < 2; ++p) {
    int j = t + p * 128;
    float v = -3.0e38f;
    if (j < L) {
      v = xr[j];
      if (bias) v += bias[((long)i * L + j) * biasH + h];
      if (mask) v += (mask[(long)sI * mldr + (long)j * mldj] - 1.0f) * NEGB;
    }
    vals[p] = v;
    mx = fmaxf(mx, v);
  }
  for (int o = 16; o; o >>= 1) mx = fmaxf(mx, __shfl_down(mx, o, 32));
  if ((t & 31) == 0) red[t >> 5] = mx;
  __syncthreads();
  mx = fmaxf(fmaxf(red[0], red[1]), fmaxf(red[2], red[3]));
  __syncthreads();
  float sum = 0.f;
#pragma unroll
  for (int p = 0; p < 2; ++p) {
    int j = t + p * 128;
    float e = (j < L) ? __expf(vals[p] - mx) : 0.f;
    vals[p] = e; sum += e;
  }
  for (int o = 16; o; o >>= 1) sum += __shfl_down(sum, o, 32);
  if ((t & 31) == 0) red[t >> 5] = sum;
  __syncthreads();
  sum = red[0] + red[1] + red[2] + red[3];
  float inv = 1.f / sum;
#pragma unroll
  for (int p = 0; p < 2; ++p) {
    int j = t + p * 128;
    if (j < L) xr[j] = vals[p] * inv;
  }
}

// ------------------------------ elementwise --------------------------------
#define EWIDX long i = (long)blockIdx.x * 256 + threadIdx.x; if (i >= n) return;

__global__ void ew_mul_k(float* __restrict__ a, const float* __restrict__ b, long n)
{ EWIDX a[i] *= b[i]; }

__global__ void ew_muladd_k(float* __restrict__ d, const float* __restrict__ a,
                            const float* __restrict__ b, long n)
{ EWIDX d[i] += a[i] * b[i]; }

__global__ void ew_mul3mask_k(float* __restrict__ d, const float* __restrict__ x,
                              const float* __restrict__ y, const float* __restrict__ m,
                              int C, long n)
{ EWIDX d[i] = x[i] * y[i] * m[i / C]; }

__global__ void ew_maskmul_k(float* __restrict__ d, const float* __restrict__ m, int C, long n)
{ EWIDX d[i] *= m[i / C]; }

__global__ void ew_masked_add_k(float* __restrict__ d, const float* __restrict__ s,
                                const float* __restrict__ m, int C, long n)
{ EWIDX d[i] += s[i] * m[i / C]; }

__global__ void ew_add_divnorm_k(float* __restrict__ d, const float* __restrict__ s,
                                 const float* __restrict__ nrm, int C, long n)
{ EWIDX d[i] += s[i] / nrm[i / C]; }

// dst_f16[c*R + r] = src[r*C + c]   (move last dim to front, convert)
__global__ void ew_lastfront_h_k(h16* __restrict__ d, const float* __restrict__ s,
                                 long R, int C)
{ long n = R * C; EWIDX long r = i / C; int c = (int)(i - r * C);
  ((_Float16*)d)[(long)c * R + r] = (_Float16)s[i]; }

// dst_f16[c*NN + p*Nn + q] = src[(q*Nn + p)*C + c]  (transposed variant)
__global__ void ew_lastfrontT_h_k(h16* __restrict__ d, const float* __restrict__ s,
                                  int Nn, int C)
{
  long n = (long)Nn * Nn * C; EWIDX
  int q = (int)(i % Nn);
  long t = i / Nn;
  int p = (int)(t % Nn);
  int c = (int)(t / Nn);
  ((_Float16*)d)[i] = (_Float16)s[(((long)q * Nn + p) * C) + c];
}

// dst[r*C + c] = src[c*R + r]
__global__ void ew_frontlast_k(float* __restrict__ d, const float* __restrict__ s,
                               long R, int C)
{ long n = R * C; EWIDX long r = i / C; int c = (int)(i - r * C); d[i] = s[(long)c * R + r]; }

// OPM chunk permute: src[il][c][j][d] -> dst[il][j][c][d], dims (CH,32,N_,32)
__global__ void ew_opmperm_k(float* __restrict__ d, const float* __restrict__ s, long n)
{
  EWIDX
  int dd = (int)(i & 31);
  long t = i >> 5;
  int j  = (int)(t % N_);
  long t2 = t / N_;
  int c  = (int)(t2 & 31);
  int il = (int)(t2 >> 5);
  d[(((long)il * N_ + j) * CO + c) * CO + dd] = s[i];
}

// dst[(a*Nn+b)*C+c] (=|+=) src[(b*Nn+a)*C+c]
__global__ void ew_transpose_cm_k(float* __restrict__ d, const float* __restrict__ s,
                                  int Nn, int C, int add)
{
  long n = (long)Nn * Nn * C; EWIDX
  int c = (int)(i % C);
  long r = i / C;
  int b2 = (int)(r % Nn);
  int a  = (int)(r / Nn);
  float v = s[(((long)b2 * Nn + a) * C) + c];
  if (add) d[i] += v; else d[i] = v;
}

__global__ void ew_transpose2d_k(float* __restrict__ d, const float* __restrict__ s, int Nn)
{ long n = (long)Nn * Nn; EWIDX d[i] = s[(i % Nn) * Nn + i / Nn]; }

// norm[i,j] = sum_s mask[s,i]*mask[s,j] + 1e-3
__global__ void opm_norm_k(const float* __restrict__ mask, float* __restrict__ nrm)
{
  long n = (long)N_ * N_; EWIDX
  int ii = (int)(i / N_), jj = (int)(i % N_);
  float s = 0.f;
  for (int t = 0; t < S_; ++t) s += mask[t * N_ + ii] * mask[t * N_ + jj];
  nrm[i] = s + 1e-3f;
}

// f32 KxN -> f16 NxK (weight / operand conversion+transpose)
__global__ void wcvt_k(const float* __restrict__ Wf, h16* __restrict__ Wt,
                       int K, int N, long n)
{
  EWIDX
  int nn = (int)(i % N);
  long k = i / N;
  ((_Float16*)Wt)[(long)nn * K + k] = (_Float16)Wf[i];
}

// vt[((o*Hh + h)*32 + d)*L + j] = v[o*vo + j*vj + h*32 + d]   (f16 V operand)
__global__ void permvt_k(const float* __restrict__ v, h16* __restrict__ vt,
                         int Hh, int L, long vo, long vj, long n)
{
  EWIDX
  int j = (int)(i % L);
  long t = i / L;
  int d = (int)(t & 31); t >>= 5;
  int h = (int)(t % Hh);
  int o = (int)(t / Hh);
  ((_Float16*)vt)[i] = (_Float16)v[(long)o * vo + (long)j * vj + h * 32 + d];
}

// ------------------------------ host side ----------------------------------
typedef struct {
  hipStream_t st;
  const float* const* PP;
  h16** WH;
  float *bMN, *bQ, *bK, *bV, *bG, *bL, *bO, *bT1, *bT2, *bPB, *bTB, *bNORM, *bPMT;
  h16 *bHA0, *bHA1, *bHA2;   // f16 activation scratch (BIG halves each)
} Ctx;

#define EWG(n) dim3((unsigned)(((n) + 255) / 256)), dim3(256)

static inline void launch_gemm(hipStream_t st, int af16,
    const float* A, const h16* Ah, const h16* Bh, const float* bias,
    float* C, h16* C16,
    int M, int N, int K, int lda, int ldb, int ldc,
    int batches, int batchH,
    long sA1, long sA2, long sB1, long sB2, long sC1, long sC2,
    float alpha, int act, int beta)
{
  int tiles = ((M + 31) / 32) * ((N + 31) / 32);
  dim3 grid((tiles + 3) / 4, batches);
  int guard = (N & 31) ? 1 : 0;
  if (af16) {
    if (guard) gemm_wmma_k<1,1><<<grid, 128, 0, st>>>(A, Ah, Bh, bias, C, C16, M, N, K,
        lda, ldb, ldc, batchH, sA1, sA2, sB1, sB2, sC1, sC2, alpha, act, beta);
    else       gemm_wmma_k<1,0><<<grid, 128, 0, st>>>(A, Ah, Bh, bias, C, C16, M, N, K,
        lda, ldb, ldc, batchH, sA1, sA2, sB1, sB2, sC1, sC2, alpha, act, beta);
  } else {
    if (guard) gemm_wmma_k<0,1><<<grid, 128, 0, st>>>(A, Ah, Bh, bias, C, C16, M, N, K,
        lda, ldb, ldc, batchH, sA1, sA2, sB1, sB2, sC1, sC2, alpha, act, beta);
    else       gemm_wmma_k<0,0><<<grid, 128, 0, st>>>(A, Ah, Bh, bias, C, C16, M, N, K,
        lda, ldb, ldc, batchH, sA1, sA2, sB1, sB2, sC1, sC2, alpha, act, beta);
  }
}

// weight GEMM (f32 A, pre-converted f16 NxK weight B), optional f16 mirror out
static inline void g1m(Ctx& c, const float* A, int widx, int biasIdx,
                       float* C, h16* C16, long M, int N, int K, int lda, int ldc,
                       float alpha, int act, int beta)
{
  launch_gemm(c.st, 0, A, nullptr, c.WH[widx],
              biasIdx >= 0 ? c.PP[biasIdx] : nullptr, C, C16,
              (int)M, N, K, lda, K, ldc, 1, 1, 0, 0, 0, 0, 0, 0, alpha, act, beta);
}
static inline void g1(Ctx& c, const float* A, int widx, int biasIdx, float* C,
                      long M, int N, int K, int lda, int ldc,
                      float alpha, int act, int beta)
{ g1m(c, A, widx, biasIdx, C, nullptr, M, N, K, lda, ldc, alpha, act, beta); }

static inline void ln_launch(hipStream_t st, const float* x, float* y,
                             const float* g, const float* b, long rows, int C)
{ ln_k<<<(unsigned)rows, 128, 0, st>>>(x, y, g, b, C); }

static inline void ew_mul(hipStream_t st, float* a, const float* b, long n)
{ ew_mul_k<<<EWG(n), 0, st>>>(a, b, n); }
static inline void ew_muladd(hipStream_t st, float* d, const float* a, const float* b, long n)
{ ew_muladd_k<<<EWG(n), 0, st>>>(d, a, b, n); }
static inline void ew_mul3mask(hipStream_t st, float* d, const float* x, const float* y,
                               const float* m, int C, long n)
{ ew_mul3mask_k<<<EWG(n), 0, st>>>(d, x, y, m, C, n); }
static inline void ew_maskmul(hipStream_t st, float* d, const float* m, int C, long n)
{ ew_maskmul_k<<<EWG(n), 0, st>>>(d, m, C, n); }
static inline void ew_masked_add(hipStream_t st, float* d, const float* s,
                                 const float* m, int C, long n)
{ ew_masked_add_k<<<EWG(n), 0, st>>>(d, s, m, C, n); }
static inline void ew_add_divnorm(hipStream_t st, float* d, const float* s,
                                  const float* nrm, int C, long n)
{ ew_add_divnorm_k<<<EWG(n), 0, st>>>(d, s, nrm, C, n); }

// Chunked multi-head attention: softmax(QK^T/sqrt(32) + bias + mask) @ V.
// qh/kh: f16 mirrors with per-outer stride sOuter, per-head +h*32, row stride ldq.
// V (f32) is re-laid out once per phase to f16 [o][h][d][j] for K-major AV loads.
static void attn(Ctx& c, const h16* qh, const h16* kh, const float* v,
                 float* o, float* logits, const float* bias, int biasH,
                 const float* mask, int mldr, int mldj, long maskCS,
                 int OUTER, int CH, int Hh, int Mq, int Lk, long sOuter, int ldq,
                 long vo, long vj)
{
  const float alpha = 0.17677669529663687f;  // 1/sqrt(32)
  long lsz = (long)Mq * Lk;
  long nvt = (long)OUTER * Hh * 32 * Lk;
  permvt_k<<<EWG(nvt), 0, c.st>>>(v, c.bHA0, Hh, Lk, vo, vj, nvt);
  for (int c0 = 0; c0 < OUTER; c0 += CH) {
    int batches = CH * Hh;
    launch_gemm(c.st, 1, nullptr, qh + (long)c0 * sOuter, kh + (long)c0 * sOuter,
                nullptr, logits, nullptr, Mq, Lk, 32, ldq, ldq, Lk, batches, Hh,
                sOuter, 32, sOuter, 32, (long)Hh * lsz, lsz, alpha, 0, 0);
    unsigned rows = (unsigned)(CH * Hh * Mq);
    softmax_k<<<rows, 128, 0, c.st>>>(logits, Lk, Mq, Hh, bias, biasH,
                                      mask + (long)c0 * maskCS, mldr, mldj);
    launch_gemm(c.st, 0, logits, nullptr, c.bHA0 + (long)c0 * Hh * 32 * Lk, nullptr,
                o + (long)c0 * sOuter, nullptr, Mq, 32, Lk, Lk, Lk, ldq, batches, Hh,
                (long)Hh * lsz, lsz, (long)Hh * 32 * Lk, (long)32 * Lk,
                sOuter, 32, 1.f, 0, 0);
  }
}

static void transition_phase(Ctx& c, float* X, const float* mask,
    int lng, int lnb, int w1, int b1, int w2, int b2,
    long rows, int C, int Ch, long rowChunk)
{
  ln_launch(c.st, X, c.bMN, c.PP[lng], c.PP[lnb], rows, C);
  for (long r0 = 0; r0 < rows; r0 += rowChunk) {
    g1(c, c.bMN + r0 * C, w1, b1, c.bT1, rowChunk, Ch, C, C, Ch, 1.f, 1, 0);
    g1(c, c.bT1, w2, b2, c.bT2, rowChunk, C, Ch, Ch, C, 1.f, 0, 0);
    ew_masked_add(c.st, X + r0 * C, c.bT2, mask + r0, C, rowChunk * C);
  }
}

static void tri_mul_phase(Ctx& c, int base, float* Zb, const float* pm, int outgoing)
{
  const long NN = (long)N_ * N_;
  ln_launch(c.st, Zb, c.bMN, c.PP[base + 0], c.PP[base + 1], NN, CZ);
  g1(c, c.bMN, base + 2, base + 3, c.bT1, NN, CT, CZ, CZ, CT, 1.f, 2, 0);
  g1(c, c.bMN, base + 4, base + 5, c.bT2, NN, CT, CZ, CZ, CT, 1.f, 0, 0);
  ew_mul3mask(c.st, c.bQ, c.bT1, c.bT2, pm, CT, NN * CT);           // a
  g1(c, c.bMN, base + 6, base + 7, c.bT1, NN, CT, CZ, CZ, CT, 1.f, 2, 0);
  g1(c, c.bMN, base + 8, base + 9, c.bT2, NN, CT, CZ, CZ, CT, 1.f, 0, 0);
  ew_mul3mask(c.st, c.bK, c.bT1, c.bT2, pm, CT, NN * CT);           // b
  if (outgoing) {   // a_t[c][i][k]=a[i,k,c], b_t[c][j][k]=b[j,k,c]  (f16)
    ew_lastfront_h_k<<<EWG(NN * CT), 0, c.st>>>(c.bHA1, c.bQ, NN, CT);
    ew_lastfront_h_k<<<EWG(NN * CT), 0, c.st>>>(c.bHA2, c.bK, NN, CT);
  } else {          // a_t[c][i][k]=a[k,i,c], b_t[c][j][k]=b[k,j,c]  (f16)
    ew_lastfrontT_h_k<<<EWG(NN * CT), 0, c.st>>>(c.bHA1, c.bQ, N_, CT);
    ew_lastfrontT_h_k<<<EWG(NN * CT), 0, c.st>>>(c.bHA2, c.bK, N_, CT);
  }
  // x_c[i,j] = sum_k a_t[c][i][k] * b_t[c][j][k]   (batched over c, all-f16)
  launch_gemm(c.st, 1, nullptr, c.bHA1, c.bHA2, nullptr, c.bL, nullptr,
              N_, N_, N_, N_, N_, N_, CT, 1, NN, 0, NN, 0, NN, 0, 1.f, 0, 0);
  ew_frontlast_k<<<EWG(NN * CT), 0, c.st>>>(c.bO, c.bL, NN, CT);    // x[i,j,c]
  ln_launch(c.st, c.bO, c.bO, c.PP[base + 12], c.PP[base + 13], NN, CT);
  g1(c, c.bO, base + 14, base + 15, c.bT1, NN, CZ, CT, CT, CZ, 1.f, 0, 0);
  g1(c, c.bMN, base + 10, base + 11, c.bT2, NN, CZ, CZ, CZ, CZ, 1.f, 2, 0);
  ew_muladd(c.st, Zb, c.bT1, c.bT2, NN * CZ);                       // z += g*out
}

static void tri_attn_phase(Ctx& c, int base, const float* Zin, const float* pm,
                           float* proj_out, int beta)
{
  const long NN = (long)N_ * N_;
  ln_launch(c.st, Zin, c.bMN, c.PP[base + 0], c.PP[base + 1], NN, CZ);
  g1m(c, c.bMN, base + 2, -1, c.bQ, c.bHA1, NN, CZ, CZ, CZ, CZ, 1.f, 0, 0);
  g1m(c, c.bMN, base + 3, -1, c.bK, c.bHA2, NN, CZ, CZ, CZ, CZ, 1.f, 0, 0);
  g1(c, c.bMN, base + 4, -1, c.bV, NN, CZ, CZ, CZ, CZ, 1.f, 0, 0);
  g1(c, c.bMN, base + 5, base + 6, c.bG, NN, CZ, CZ, CZ, CZ, 1.f, 2, 0);
  g1(c, c.bMN, base + 9, -1, c.bTB, NN, HP, CZ, CZ, HP, 1.f, 0, 0);
  attn(c, c.bHA1, c.bHA2, c.bV, c.bO, c.bL, c.bTB, HP, pm, N_, 1, (long)N_,
       N_, 32, HP, N_, N_, (long)N_ * CZ, CZ, (long)N_ * CZ, (long)CZ);
  ew_mul(c.st, c.bO, c.bG, NN * CZ);
  g1(c, c.bO, base + 7, base + 8, proj_out, NN, CZ, CZ, CZ, CZ, 1.f, 0, beta);
}

// ---------------------------------------------------------------------------
extern "C" void kernel_launch(void* const* d_in, const int* in_sizes, int n_in,
                              void* d_out, int out_size, void* d_ws, size_t ws_size,
                              hipStream_t stream)
{
  (void)in_sizes; (void)n_in; (void)out_size; (void)ws_size;
  const float* const* PP = (const float* const*)d_in;
  const float* msa_mask  = PP[IN_MSAMASK];
  const float* pair_mask = PP[IN_PAIRMASK];

  const long BIG = (long)S_ * N_ * CM;   // 8388608 == N_*N_*CZ
  const long SN  = (long)S_ * N_;
  const long NN  = (long)N_ * N_;

  float* Mb = (float*)d_out;             // m (updated in place)
  float* Zb = Mb + BIG;                  // z (updated in place)

  float* w = (float*)d_ws;
  Ctx c;
  c.st = stream; c.PP = PP;
  c.bMN = w; w += BIG;
  c.bQ  = w; w += BIG;
  c.bK  = w; w += BIG;
  c.bV  = w; w += BIG;
  c.bG  = w; w += BIG;
  c.bL  = w; w += BIG;
  c.bO  = w; w += BIG;
  c.bT1 = w; w += BIG;
  c.bT2 = w; w += BIG;
  c.bPB   = w; w += NN * HM;
  c.bTB   = w; w += NN * HP;
  c.bNORM = w; w += NN;
  c.bPMT  = w; w += NN;

  // f16 arena: weights (NxK) first, then three activation scratch buffers
  h16* harena = (h16*)w;
  h16* WH[100] = {};
  c.WH = WH;
  size_t hoff = 0;
  auto wc = [&](int idx, int K, int N) {
    WH[idx] = harena + hoff;
    hoff += (((size_t)K * N + 7) & ~(size_t)7);
    long n = (long)K * N;
    wcvt_k<<<EWG(n), 0, stream>>>(PP[idx], WH[idx], K, N, n);
  };
  wc(R_QW, CM, HM * DM); wc(R_KW, CM, HM * DM); wc(R_VW, CM, HM * DM);
  wc(R_GW, CM, HM * DM); wc(R_OW, HM * DM, CM); wc(R_BW, CZ, HM);
  wc(C_QW, CM, HM * DM); wc(C_KW, CM, HM * DM); wc(C_VW, CM, HM * DM);
  wc(C_GW, CM, HM * DM); wc(C_OW, HM * DM, CM);
  wc(MT_L1W, CM, 4 * CM); wc(MT_L2W, 4 * CM, CM);
  wc(OP_AW, CM, CO); wc(OP_BW, CM, CO); wc(OP_OW, CO * CO, CZ);
  for (int base = TMO_BASE; base <= TMI_BASE; base += 16) {
    wc(base + 2, CZ, CT); wc(base + 4, CZ, CT); wc(base + 6, CZ, CT);
    wc(base + 8, CZ, CT); wc(base + 10, CZ, CZ); wc(base + 14, CT, CZ);
  }
  for (int base = TAS_BASE; base <= TAE_BASE; base += 10) {
    wc(base + 2, CZ, CZ); wc(base + 3, CZ, CZ); wc(base + 4, CZ, CZ);
    wc(base + 5, CZ, CZ); wc(base + 7, CZ, CZ); wc(base + 9, CZ, HP);
  }
  wc(PT_L1W, CZ, 4 * CZ); wc(PT_L2W, 4 * CZ, CZ);
  c.bHA0 = harena + hoff;
  c.bHA1 = c.bHA0 + BIG;
  c.bHA2 = c.bHA1 + BIG;

  hipMemcpyAsync(Mb, PP[IN_M], BIG * sizeof(float), hipMemcpyDeviceToDevice, stream);
  hipMemcpyAsync(Zb, PP[IN_Z], BIG * sizeof(float), hipMemcpyDeviceToDevice, stream);

  // ---------------- 1. MSA row attention (pair bias, gated) ----------------
  ln_launch(stream, Mb, c.bMN, PP[R_LN_G], PP[R_LN_B], SN, CM);
  ln_launch(stream, Zb, c.bT1, PP[R_LNZ_G], PP[R_LNZ_B], NN, CZ);
  g1(c, c.bT1, R_BW, -1, c.bPB, NN, HM, CZ, CZ, HM, 1.f, 0, 0);
  g1m(c, c.bMN, R_QW, -1, c.bQ, c.bHA1, SN, HM * DM, CM, CM, HM * DM, 1.f, 0, 0);
  g1m(c, c.bMN, R_KW, -1, c.bK, c.bHA2, SN, HM * DM, CM, CM, HM * DM, 1.f, 0, 0);
  g1(c, c.bMN, R_VW, -1, c.bV, SN, HM * DM, CM, CM, HM * DM, 1.f, 0, 0);
  g1(c, c.bMN, R_GW, R_GB, c.bG, SN, HM * DM, CM, CM, HM * DM, 1.f, 2, 0);
  attn(c, c.bHA1, c.bHA2, c.bV, c.bO, c.bL, c.bPB, HM, msa_mask, N_, 1, (long)N_,
       S_, 16, HM, N_, N_, (long)N_ * CM, CM, (long)N_ * CM, (long)CM);
  ew_mul(stream, c.bO, c.bG, SN * HM * DM);
  g1(c, c.bO, R_OW, R_OB, Mb, SN, CM, HM * DM, HM * DM, CM, 1.f, 0, 1);

  // ---------------- 2. MSA column attention (gated) ------------------------
  ln_launch(stream, Mb, c.bMN, PP[C_LN_G], PP[C_LN_B], SN, CM);
  g1m(c, c.bMN, C_QW, -1, c.bQ, c.bHA1, SN, HM * DM, CM, CM, HM * DM, 1.f, 0, 0);
  g1m(c, c.bMN, C_KW, -1, c.bK, c.bHA2, SN, HM * DM, CM, CM, HM * DM, 1.f, 0, 0);
  g1(c, c.bMN, C_VW, -1, c.bV, SN, HM * DM, CM, CM, HM * DM, 1.f, 0, 0);
  g1(c, c.bMN, C_GW, C_GB, c.bG, SN, HM * DM, CM, CM, HM * DM, 1.f, 2, 0);
  attn(c, c.bHA1, c.bHA2, c.bV, c.bO, c.bL, nullptr, 1, msa_mask, 1, N_, 1,
       N_, 64, HM, S_, S_, (long)CM, N_ * CM, (long)CM, (long)N_ * CM);
  ew_mul(stream, c.bO, c.bG, SN * HM * DM);
  g1(c, c.bO, C_OW, C_OB, Mb, SN, CM, HM * DM, HM * DM, CM, 1.f, 0, 1);

  // ---------------- 3. MSA transition --------------------------------------
  transition_phase(c, Mb, msa_mask, MT_LN_G, MT_LN_B, MT_L1W, MT_L1B,
                   MT_L2W, MT_L2B, SN, CM, 4 * CM, 8192);

  // ---------------- 4. Outer product mean ----------------------------------
  ln_launch(stream, Mb, c.bMN, PP[OP_LN_G], PP[OP_LN_B], SN, CM);
  g1(c, c.bMN, OP_AW, OP_AB, c.bQ, SN, CO, CM, CM, CO, 1.f, 0, 0);
  ew_maskmul(stream, c.bQ, msa_mask, CO, SN * CO);
  g1(c, c.bMN, OP_BW, OP_BB, c.bK, SN, CO, CM, CM, CO, 1.f, 0, 0);
  ew_maskmul(stream, c.bK, msa_mask, CO, SN * CO);
  { long n = (long)S_ * N_ * CO;   // a,b: f32 [S,8192] -> f16 [8192,S]
    wcvt_k<<<EWG(n), 0, stream>>>(c.bQ, c.bHA1, S_, N_ * CO, n);
    wcvt_k<<<EWG(n), 0, stream>>>(c.bK, c.bHA2, S_, N_ * CO, n); }
  opm_norm_k<<<EWG(NN), 0, stream>>>(msa_mask, c.bNORM);
  for (int i0 = 0; i0 < N_; i0 += 32) {
    // outer[(i,c),(j,d)] = sum_s a[s,i,c] b[s,j,d]   (all-f16 operands)
    launch_gemm(stream, 1, nullptr, c.bHA1 + (long)i0 * CO * S_, c.bHA2,
                nullptr, c.bL, nullptr, 32 * CO, N_ * CO, S_, S_, S_, N_ * CO,
                1, 1, 0, 0, 0, 0, 0, 0, 1.f, 0, 0);
    ew_opmperm_k<<<EWG(BIG), 0, stream>>>(c.bO, c.bL, BIG);   // -> [(i,j),(c,d)]
    g1(c, c.bO, OP_OW, OP_OB, c.bT2, 32L * N_, CZ, CO * CO, CO * CO, CZ, 1.f, 0, 0);
    ew_add_divnorm(stream, Zb + (long)i0 * N_ * CZ, c.bT2,
                   c.bNORM + (long)i0 * N_, CZ, 32L * N_ * CZ);
  }

  // ---------------- 5/6. Triangle multiplication out/in --------------------
  tri_mul_phase(c, TMO_BASE, Zb, pair_mask, 1);
  tri_mul_phase(c, TMI_BASE, Zb, pair_mask, 0);

  // ---------------- 7. Triangle attention (starting node) ------------------
  tri_attn_phase(c, TAS_BASE, Zb, pair_mask, Zb, 1);

  // ---------------- 8. Triangle attention (ending node, via transpose) -----
  ew_transpose_cm_k<<<EWG(NN * CZ), 0, stream>>>(c.bT1, Zb, N_, CZ, 0);   // zt
  ew_transpose2d_k<<<EWG(NN), 0, stream>>>(c.bPMT, pair_mask, N_);        // mask^T
  tri_attn_phase(c, TAE_BASE, c.bT1, c.bPMT, c.bT1, 0);
  ew_transpose_cm_k<<<EWG(NN * CZ), 0, stream>>>(Zb, c.bT1, N_, CZ, 1);   // z += out^T

  // ---------------- 9. Pair transition -------------------------------------
  transition_phase(c, Zb, pair_mask, PT_LN_G, PT_LN_B, PT_L1W, PT_L1B,
                   PT_L2W, PT_L2B, NN, CZ, 4 * CZ, 16384);
}
